// Caser_22668837388599
// MI455X (gfx1250) — matrix operand order repaired and checked by reading, not compiled, not run on hardware
//
#include <hip/hip_runtime.h>

// ---------------------------------------------------------------------------
// Caser forward for MI455X (gfx1250, wave32, WMMA).
// Dominant compute (46 GFLOP of horizontal convs) runs on
// v_wmma_f32_16x16x32_bf16 with M=16 filters, N=16 batch rows, K=h*64
// (multiple of 64 -> K-loop unrolled x2, no masking). Two output positions
// are processed per wave pass so each A fragment feeds 2 WMMAs.
// Dominant memory (512MB W2 gather) stays VALU/float4 in the 192MB L2.
// ---------------------------------------------------------------------------

typedef __attribute__((ext_vector_type(16))) __bf16        v16bf;
typedef __attribute__((ext_vector_type(8)))  float         v8f;

#define B_      1024
#define L_      50
#define E_      64
#define FV_     4
#define FH_     16
#define T_      1024
#define XROW    (L_ * E_)      // 3200 bf16 elements per batch row
#define XROW_P  (XROW + 8)     // padded LDS row to break bank conflicts
#define CONCAT  1056           // FV*E + FH*L = 256 + 800
#define NGRP    5              // filter-length groups (i round-robin mod 5)

__device__ __forceinline__ unsigned short f2bf(float f) {
  unsigned u = __float_as_uint(f);
  u = (u + 0x7FFFu + ((u >> 16) & 1u)) >> 16;   // round-to-nearest-even
  return (unsigned short)u;
}

// Build a 16-element bf16 fragment from two 16B chunks; bit_cast keeps the
// loads in the consecutive even-aligned VGPRs the WMMA operands require.
__device__ __forceinline__ v16bf ld_frag2(const unsigned short* p0,
                                          const unsigned short* p1) {
  struct alignas(16) Pair { uint4 a, b; } pr;
  pr.a = *(const uint4*)p0;
  pr.b = *(const uint4*)p1;
  return __builtin_bit_cast(v16bf, pr);
}

__device__ __forceinline__ v8f wmma_bf16(v16bf a, v16bf b, v8f c) {
  return __builtin_amdgcn_wmma_f32_16x16x32_bf16(false, a, false, b,
                                                 (short)0, c, false, false);
}

// --------------------------------------------------------------------------
// K1: gather item embeddings (f32 -> bf16 rows) + vertical conv out_v.
// grid = B, block = 64 (thread = e).
// --------------------------------------------------------------------------
__global__ void gather_outv_kernel(const int* __restrict__ seqs,
                                   const float* __restrict__ item_table,
                                   const float* __restrict__ Wv,
                                   const float* __restrict__ bv,
                                   unsigned short* __restrict__ Xbf,
                                   float* __restrict__ out_concat) {
  int b = blockIdx.x;
  int e = threadIdx.x;
  float a0 = 0.f, a1 = 0.f, a2 = 0.f, a3 = 0.f;
  for (int l = 0; l < L_; ++l) {
    int idx = seqs[b * L_ + l];
    float v = item_table[(size_t)idx * E_ + e];
    Xbf[(size_t)b * XROW + l * E_ + e] = f2bf(v);
    a0 += v * Wv[0 * L_ + l];
    a1 += v * Wv[1 * L_ + l];
    a2 += v * Wv[2 * L_ + l];
    a3 += v * Wv[3 * L_ + l];
  }
  float* o = out_concat + (size_t)b * CONCAT;   // out_v occupies [0,256)
  o[0 * E_ + e] = a0 + bv[0];
  o[1 * E_ + e] = a1 + bv[1];
  o[2 * E_ + e] = a2 + bv[2];
  o[3 * E_ + e] = a3 + bv[3];
}

// --------------------------------------------------------------------------
// K1b: convert Wh (50*16*50*64 f32) to bf16.
// --------------------------------------------------------------------------
__global__ void cvt_wh_kernel(const float* __restrict__ Wh,
                              unsigned short* __restrict__ Whbf, int n) {
  int i = blockIdx.x * blockDim.x + threadIdx.x;
  int stride = gridDim.x * blockDim.x;
  for (; i < n; i += stride) Whbf[i] = f2bf(Wh[i]);
}

// --------------------------------------------------------------------------
// K2: horizontal convs via WMMA.
//   grid = (B/16, NGRP), block = 256 (8 waves).
//   A = Wh[i] as 16x(h*64) bf16 (global, L2-resident, per-lane layout loads)
//   B = 16 batch rows' contiguous windows from LDS
//   Two t positions per wave pass share A fragments; K-loop unrolled x2
//   (K = h*64 is always a multiple of 64) -> 4 WMMAs per inner iteration.
//   C = f32; relu+bias+max(t) in registers, cross-wave max via atomicMax on
//   non-negative float bit patterns in LDS.
// Dynamic LDS = 16*XROW_P*2 + 256*4 = 103,680 B (<< 320 KB/WG on CDNA5).
// --------------------------------------------------------------------------
__global__ void conv_wmma_kernel(const unsigned short* __restrict__ Xbf,
                                 const unsigned short* __restrict__ Whbf,
                                 const float* __restrict__ bh,
                                 float* __restrict__ out_concat) {
  extern __shared__ char smem[];
  unsigned short* Xlds = (unsigned short*)smem;                       // 16 rows
  int* maxbuf = (int*)(smem + (size_t)16 * XROW_P * sizeof(unsigned short));

  const int b0   = blockIdx.x * 16;
  const int grp  = blockIdx.y;
  const int tid  = threadIdx.x;
  const int wave = tid >> 5;
  const int lane = tid & 31;
  const int n    = lane & 15;   // batch column (B/C) and filter row (A)
  const int half = lane >> 4;

  // Cooperative copy: 16 contiguous batch rows -> padded LDS rows.
  {
    const uint4* src = (const uint4*)(Xbf + (size_t)b0 * XROW);
    for (int idx = tid; idx < 16 * (XROW / 8); idx += 256) {
      int row = idx / (XROW / 8);
      int col = idx % (XROW / 8);
      *(uint4*)&Xlds[row * XROW_P + col * 8] = src[idx];
    }
  }
  maxbuf[tid] = 0;   // relu output >= 0, so 0 is the identity for the max
  __syncthreads();

  for (int i = grp; i < L_; i += NGRP) {
    const int h    = i + 1;
    const int Tpos = L_ - h + 1;
    const int K    = h * E_;                       // multiple of 64 always
    const unsigned short* WA =
        Whbf + (size_t)i * (FH_ * XROW) + (size_t)n * XROW + 16 * half;

    float bias[8];
#pragma unroll
    for (int r = 0; r < 8; ++r) bias[r] = bh[i * FH_ + 8 * half + r];

    float acc[8];
#pragma unroll
    for (int r = 0; r < 8; ++r) acc[r] = 0.f;

    // Wave-uniform loop: EXEC stays all-ones through every WMMA.
    for (int u = wave; 2 * u < Tpos; u += 8) {
      const int t0 = 2 * u;
      const unsigned short* brow = Xlds + n * XROW_P + t0 * E_ + 16 * half;
      v8f c0 = {};
      if (t0 + 1 < Tpos) {                          // paired: A reused 2x
        v8f c1 = {};
        for (int k0 = 0; k0 < K; k0 += 64) {
          v16bf a0 = ld_frag2(WA + k0,        WA + k0 + 8);
          v16bf a1 = ld_frag2(WA + k0 + 32,   WA + k0 + 40);
          v16bf b00 = ld_frag2(brow + k0,          brow + k0 + 8);
          v16bf b01 = ld_frag2(brow + k0 + 32,     brow + k0 + 40);
          v16bf b10 = ld_frag2(brow + 64 + k0,     brow + 64 + k0 + 8);
          v16bf b11 = ld_frag2(brow + 64 + k0 + 32, brow + 64 + k0 + 40);
          c0 = wmma_bf16(a0, b00, c0);
          c0 = wmma_bf16(a1, b01, c0);
          c1 = wmma_bf16(a0, b10, c1);
          c1 = wmma_bf16(a1, b11, c1);
        }
#pragma unroll
        for (int r = 0; r < 8; ++r) {
          float v = c1[r] + bias[r];
          v = v > 0.f ? v : 0.f;
          acc[r] = acc[r] > v ? acc[r] : v;
        }
      } else {                                      // last odd position
        for (int k0 = 0; k0 < K; k0 += 64) {
          v16bf a0 = ld_frag2(WA + k0,      WA + k0 + 8);
          v16bf a1 = ld_frag2(WA + k0 + 32, WA + k0 + 40);
          v16bf b00 = ld_frag2(brow + k0,      brow + k0 + 8);
          v16bf b01 = ld_frag2(brow + k0 + 32, brow + k0 + 40);
          c0 = wmma_bf16(a0, b00, c0);
          c0 = wmma_bf16(a1, b01, c0);
        }
      }
#pragma unroll
      for (int r = 0; r < 8; ++r) {                // C: lane holds (f=r+8*half, n)
        float v = c0[r] + bias[r];
        v = v > 0.f ? v : 0.f;
        acc[r] = acc[r] > v ? acc[r] : v;
      }
    }
#pragma unroll
    for (int r = 0; r < 8; ++r)
      atomicMax(&maxbuf[n * 16 + 8 * half + r], __float_as_int(acc[r]));
    __syncthreads();
    {
      int bl = tid >> 4, f = tid & 15;
      out_concat[(size_t)(b0 + bl) * CONCAT + 256 + i * 16 + f] =
          __int_as_float(maxbuf[tid]);
      maxbuf[tid] = 0;
    }
    __syncthreads();
  }
}

// --------------------------------------------------------------------------
// K3: z = relu(out @ W1^T + b1); x = concat(z, user_emb).
// grid = B, block = 64 (thread = output channel e).
// --------------------------------------------------------------------------
__global__ void mlp_kernel(const float* __restrict__ out_concat,
                           const float* __restrict__ W1,
                           const float* __restrict__ b1,
                           const int* __restrict__ user_var,
                           const float* __restrict__ user_table,
                           float* __restrict__ x_vec) {
  __shared__ float row[CONCAT];
  int b = blockIdx.x, e = threadIdx.x;
  for (int k = e; k < CONCAT; k += 64) row[k] = out_concat[(size_t)b * CONCAT + k];
  __syncthreads();
  const float4* w  = (const float4*)(W1 + (size_t)e * CONCAT);
  const float4* r4 = (const float4*)row;
  float s = b1[e];
  for (int k = 0; k < CONCAT / 4; ++k) {
    float4 a = w[k], c = r4[k];
    s += a.x * c.x + a.y * c.y + a.z * c.z + a.w * c.w;
  }
  x_vec[(size_t)b * 128 + e]      = s > 0.f ? s : 0.f;
  x_vec[(size_t)b * 128 + 64 + e] = user_table[(size_t)user_var[b] * E_ + e];
}

// --------------------------------------------------------------------------
// K4: res[b,t] = dot(W2_table[idx], x[b]) + b2_table[idx].  Memory-bound;
// W2_table (~102MB) is L2-resident on the 192MB L2.  The gathered row is
// prefetched (global_prefetch_b8) so it overlaps the xs fill + barrier.
// grid = (B, T/256), block = 256.
// --------------------------------------------------------------------------
__global__ void final_kernel(const int* __restrict__ items,
                             const float* __restrict__ W2,
                             const float* __restrict__ b2,
                             const float* __restrict__ x_vec,
                             float* __restrict__ out) {
  __shared__ float xs[128];
  int b = blockIdx.x, tid = threadIdx.x;
  int t = blockIdx.y * 256 + tid;
  int idx = items[(size_t)b * T_ + t];
  const float4* w = (const float4*)(W2 + (size_t)idx * 128);
  __builtin_prefetch((const void*)w, 0, 0);     // global_prefetch_b8
  if (tid < 128) xs[tid] = x_vec[(size_t)b * 128 + tid];
  __syncthreads();
  const float4* xv = (const float4*)xs;
  float s = 0.f;
#pragma unroll
  for (int j = 0; j < 32; ++j) {
    float4 a = w[j], c = xv[j];
    s += a.x * c.x + a.y * c.y + a.z * c.z + a.w * c.w;
  }
  out[(size_t)b * T_ + t] = s + b2[idx];
}

// --------------------------------------------------------------------------
extern "C" void kernel_launch(void* const* d_in, const int* in_sizes, int n_in,
                              void* d_out, int out_size, void* d_ws, size_t ws_size,
                              hipStream_t stream) {
  const int*   seqs  = (const int*)  d_in[0];   // (B, L)
  const int*   items = (const int*)  d_in[1];   // (B, 1, T)
  const int*   user  = (const int*)  d_in[2];   // (B,)
  const float* utab  = (const float*)d_in[3];   // (NU+1, E)
  const float* itab  = (const float*)d_in[4];   // (NI+1, E)
  const float* Wv    = (const float*)d_in[5];   // (4,1,50,1)
  const float* bv    = (const float*)d_in[6];   // (4,)
  const float* Wh    = (const float*)d_in[7];   // (50,16,50,64)
  const float* bh    = (const float*)d_in[8];   // (50,16)
  const float* W1    = (const float*)d_in[9];   // (64,1056)
  const float* b1    = (const float*)d_in[10];  // (64,)
  const float* W2    = (const float*)d_in[11];  // (NI+1, 128)
  const float* b2    = (const float*)d_in[12];  // (NI+1, 1)
  float* out = (float*)d_out;

  // Workspace layout (16.5 MB total, 16B-aligned sections).
  char* ws = (char*)d_ws;
  unsigned short* Xbf   = (unsigned short*)(ws);             // B*3200 bf16 = 6,553,600 B
  unsigned short* Whbf  = (unsigned short*)(ws + 6553600);   // 2.56M bf16  = 5,120,000 B
  float*          oconc = (float*)(ws + 11673600);           // B*1056 f32  = 4,325,376 B
  float*          x_vec = (float*)(ws + 15998976);           // B*128 f32   =   524,288 B

  gather_outv_kernel<<<dim3(B_), dim3(E_), 0, stream>>>(seqs, itab, Wv, bv, Xbf, oconc);
  cvt_wh_kernel<<<dim3(4096), dim3(256), 0, stream>>>(Wh, Whbf, L_ * FH_ * L_ * E_);

  size_t smem = (size_t)16 * XROW_P * sizeof(unsigned short) + 256 * sizeof(int);
  conv_wmma_kernel<<<dim3(B_ / 16, NGRP), dim3(256), smem, stream>>>(Xbf, Whbf, bh, oconc);

  mlp_kernel<<<dim3(B_), dim3(E_), 0, stream>>>(oconc, W1, b1, user, utab, x_vec);
  final_kernel<<<dim3(B_, T_ / 256), dim3(256), 0, stream>>>(items, W2, b2, x_vec, out);
}
